// UPFDSingle_1219770712147
// MI455X (gfx1250) — matrix operand order repaired and checked by reading, not compiled, not run on hardware
//
#include <hip/hip_runtime.h>
#include <hip/hip_bf16.h>

// ---------------------------------------------------------------------------
// UPFD-style GCN forward: GCNConv(768->128, sym-norm, self-loops) + ReLU,
// global mean pool over 128 graphs, MLP head 128->128->2, log_softmax.
//
// GEMM (x @ W_gcn) runs on v_wmma_f32_16x16x32_bf16 using a bf16 hi/lo split
// (3 WMMAs per K-step) for ~fp32 accuracy. A-tile is converted once per block
// into LDS (fragment layout) and consumed via ds_load_b128; B fragments are
// double-buffered from an L2-resident global pack. Everything else is
// bandwidth / atomic bound (coalesced loads + global_atomic_add_f32).
// ---------------------------------------------------------------------------

#define N_NODES  50000
#define N_EDGES  500000
#define N_GRAPHS 128
#define IN_DIM   768
#define HIDDEN   128
#define OUT_DIM  2

#define M_TILES  (N_NODES / 16)   // 3125 (exact)
#define N_TILES  (HIDDEN / 16)    // 8
#define K_TILES  (IN_DIM / 32)    // 24

typedef __attribute__((ext_vector_type(16))) __bf16 v16bf;
typedef __attribute__((ext_vector_type(8)))  float  v8f;

// ---- fp32 <-> bf16 helpers (round-to-nearest-even) -------------------------
static __device__ __forceinline__ unsigned short f2bf(float f) {
    unsigned int u = __float_as_uint(f);
    u += 0x7FFFu + ((u >> 16) & 1u);
    return (unsigned short)(u >> 16);
}
static __device__ __forceinline__ float bf2f(unsigned short h) {
    return __uint_as_float(((unsigned int)h) << 16);
}
static __device__ __forceinline__ v16bf as_bf16x16(const unsigned int* u) {
    union { unsigned int u[8]; v16bf v; } c;
#pragma unroll
    for (int i = 0; i < 8; ++i) c.u[i] = u[i];
    return c.v;
}

// ---------------------------------------------------------------------------
// K0: init workspace (deg=1 for self loop, acc/gsum/gcount = 0)
// ---------------------------------------------------------------------------
__global__ void k_init(unsigned int* __restrict__ deg, float* __restrict__ acc,
                       float* __restrict__ gsum, float* __restrict__ gcount) {
    int i = blockIdx.x * 256 + threadIdx.x;
    if (i < N_NODES * HIDDEN) acc[i] = 0.0f;      // 6.4M
    if (i < N_NODES)          deg[i] = 1u;        // self-loop
    if (i < N_GRAPHS * HIDDEN) gsum[i] = 0.0f;
    if (i < N_GRAPHS)          gcount[i] = 0.0f;
}

// ---------------------------------------------------------------------------
// K1: in-degree over real edges (self loop already counted via init=1)
// ---------------------------------------------------------------------------
__global__ void k_degree(const int* __restrict__ ei, unsigned int* __restrict__ deg) {
    int e = blockIdx.x * 256 + threadIdx.x;
    if (e < N_EDGES) atomicAdd(&deg[ei[N_EDGES + e]], 1u);   // dst row
}

// ---------------------------------------------------------------------------
// K2: deg^{-1/2}  (deg >= 1 always)
// ---------------------------------------------------------------------------
__global__ void k_dinv(const unsigned int* __restrict__ deg, float* __restrict__ dinv) {
    int i = blockIdx.x * 256 + threadIdx.x;
    if (i < N_NODES) dinv[i] = rsqrtf((float)deg[i]);
}

// ---------------------------------------------------------------------------
// K3: pack W_gcn (768x128 f32, row-major) into bf16 hi/lo WMMA-B fragments.
// B 32x16 bf16 layout (wave32): lane = ((K>>3)&1)*16+N, vgpr j = ((K>>4)<<2)+((K&7)>>1),
// 16-bit half = K&1. Fragment-contiguous: dword idx = ((kt*8+nt)*32+lane)*8+j
// ---------------------------------------------------------------------------
__global__ void k_packW(const float* __restrict__ W,
                        unsigned short* __restrict__ whi,
                        unsigned short* __restrict__ wlo) {
    int id = blockIdx.x * 256 + threadIdx.x;
    if (id >= IN_DIM * HIDDEN) return;
    int k = id / HIDDEN, n = id % HIDDEN;
    float f = W[id];
    unsigned short hi = f2bf(f);
    unsigned short lo = f2bf(f - bf2f(hi));
    int kt = k >> 5, kk = k & 31, nt = n >> 4, nn = n & 15;
    int lane = (((kk >> 3) & 1) << 4) | nn;
    int j    = ((kk >> 4) << 2) + ((kk & 7) >> 1);
    int idx  = ((((kt * N_TILES + nt) * 32 + lane) * 8 + j) << 1) | (kk & 1);
    whi[idx] = hi;
    wlo[idx] = lo;
}

// ---------------------------------------------------------------------------
// K4: h = x @ W_gcn via WMMA bf16 (hi/lo split, f32 accumulate).
// Block = 256 threads = 8 waves; wave w owns N-tile w; blockIdx.x = M-tile.
//
// Phase 1 (whole block): load 16x768 fp32 x-tile coalesced, split to bf16
//   hi/lo ONCE, store to LDS in A-fragment layout:
//     A 16x32 bf16: element (m,kk): lane = m | ((kk>>3)&1)<<4,
//                   vgpr j = ((kk>>4)<<2) + ((kk&7)>>1), 16-bit half = kk&1.
//   LDS dword index = (kt*32 + lane)*8 + j  (hi and lo planes, 24 KB each).
// Phase 2 (per wave): ds_load_b128 A fragments, double-buffered global B
//   fragments, fully unrolled K loop -> 72 v_wmma per tile.
// ---------------------------------------------------------------------------
__global__ void __launch_bounds__(256)
k_gemm_wmma(const float* __restrict__ x,
            const unsigned int* __restrict__ whi,
            const unsigned int* __restrict__ wlo,
            float* __restrict__ h) {
    __shared__ unsigned int lAhi[K_TILES * 32 * 8];   // 24 KB
    __shared__ unsigned int lAlo[K_TILES * 32 * 8];   // 24 KB

    const int tid = threadIdx.x;
    const int mt  = blockIdx.x;

    // ---- Phase 1: cooperative A-tile conversion (16*768 = 12288 elements)
    {
        unsigned short* shi = (unsigned short*)lAhi;
        unsigned short* slo = (unsigned short*)lAlo;
        const float* xt = x + (size_t)mt * 16 * IN_DIM;
#pragma unroll
        for (int it = 0; it < (16 * IN_DIM) / 256; ++it) {
            int id = it * 256 + tid;           // coalesced over k
            int m  = id / IN_DIM;
            int k  = id % IN_DIM;
            float f = xt[id];
            unsigned short hi = f2bf(f);
            unsigned short lo = f2bf(f - bf2f(hi));
            int kt   = k >> 5, kk = k & 31;
            int lane = m | (((kk >> 3) & 1) << 4);
            int j    = ((kk >> 4) << 2) + ((kk & 7) >> 1);
            int idx  = (((kt * 32 + lane) * 8 + j) << 1) | (kk & 1);
            shi[idx] = hi;
            slo[idx] = lo;
        }
    }
    __syncthreads();

    // ---- Phase 2: per-wave WMMA over 24 K-tiles, double-buffered B
    const int lane = tid & 31;
    const int nt   = tid >> 5;                 // wave id = N tile (0..7)

    v8f acc = 0.0f;
    unsigned int bh[2][8], bl[2][8];

    // preload B for kt = 0
    {
        const unsigned int* p0 = whi + (((size_t)(0 * N_TILES + nt) * 32 + lane) << 3);
        const unsigned int* p1 = wlo + (((size_t)(0 * N_TILES + nt) * 32 + lane) << 3);
        *(uint4*)(bh[0])     = *(const uint4*)(p0);
        *(uint4*)(bh[0] + 4) = *(const uint4*)(p0 + 4);
        *(uint4*)(bl[0])     = *(const uint4*)(p1);
        *(uint4*)(bl[0] + 4) = *(const uint4*)(p1 + 4);
    }

#pragma unroll
    for (int kt = 0; kt < K_TILES; ++kt) {
        const int cur = kt & 1, nxt = cur ^ 1;
        // prefetch next B tile while this tile computes
        if (kt + 1 < K_TILES) {
            const unsigned int* p0 =
                whi + (((size_t)((kt + 1) * N_TILES + nt) * 32 + lane) << 3);
            const unsigned int* p1 =
                wlo + (((size_t)((kt + 1) * N_TILES + nt) * 32 + lane) << 3);
            *(uint4*)(bh[nxt])     = *(const uint4*)(p0);
            *(uint4*)(bh[nxt] + 4) = *(const uint4*)(p0 + 4);
            *(uint4*)(bl[nxt])     = *(const uint4*)(p1);
            *(uint4*)(bl[nxt] + 4) = *(const uint4*)(p1 + 4);
        }
        // A fragments from LDS (shared by all 8 waves)
        unsigned int ah[8], al[8];
        const unsigned int* sa = lAhi + ((kt * 32 + lane) << 3);
        const unsigned int* sb = lAlo + ((kt * 32 + lane) << 3);
        *(uint4*)(ah)     = *(const uint4*)(sa);
        *(uint4*)(ah + 4) = *(const uint4*)(sa + 4);
        *(uint4*)(al)     = *(const uint4*)(sb);
        *(uint4*)(al + 4) = *(const uint4*)(sb + 4);

        v16bf va_hi = as_bf16x16(ah);
        v16bf va_lo = as_bf16x16(al);
        v16bf vb_hi = as_bf16x16(bh[cur]);
        v16bf vb_lo = as_bf16x16(bl[cur]);

        // x@W ~= xh*Wh + xl*Wh + xh*Wl   (xl*Wl ~ 2^-16, dropped)
        acc = __builtin_amdgcn_wmma_f32_16x16x32_bf16(false, va_hi, false, vb_hi,
                                                      (short)0, acc, false, false);
        acc = __builtin_amdgcn_wmma_f32_16x16x32_bf16(false, va_lo, false, vb_hi,
                                                      (short)0, acc, false, false);
        acc = __builtin_amdgcn_wmma_f32_16x16x32_bf16(false, va_hi, false, vb_lo,
                                                      (short)0, acc, false, false);
    }

    // ---- C/D layout: vgpr r -> M = r + (lane>=16 ? 8 : 0), N = lane&15
    const int mofs = (lane >> 4) << 3;
    const int col  = nt * 16 + (lane & 15);
#pragma unroll
    for (int r = 0; r < 8; ++r)
        h[(size_t)(mt * 16 + r + mofs) * HIDDEN + col] = acc[r];
}

// ---------------------------------------------------------------------------
// K5: edge scatter: acc[dst] += h[src] * dinv[src]*dinv[dst]
// 128 threads per edge (one feature each), 2 edges per 256-thread block.
// ---------------------------------------------------------------------------
__global__ void __launch_bounds__(256)
k_scatter(const int* __restrict__ ei, const float* __restrict__ h,
          const float* __restrict__ dinv, float* __restrict__ acc) {
    int e = blockIdx.x * 2 + (threadIdx.x >> 7);
    int f = threadIdx.x & 127;
    if (e >= N_EDGES) return;
    int s = ei[e];
    int d = ei[N_EDGES + e];
    float nrm = dinv[s] * dinv[d];
    atomicAdd(acc + (size_t)d * HIDDEN + f, h[(size_t)s * HIDDEN + f] * nrm);
}

// ---------------------------------------------------------------------------
// K6: self-loop + bias + ReLU, then mean-pool accumulation per graph.
// ---------------------------------------------------------------------------
__global__ void __launch_bounds__(256)
k_relu_pool(const float* __restrict__ acc, const float* __restrict__ h,
            const float* __restrict__ dinv, const float* __restrict__ bg,
            const int* __restrict__ batch,
            float* __restrict__ gsum, float* __restrict__ gcount) {
    int i = blockIdx.x * 2 + (threadIdx.x >> 7);
    int f = threadIdx.x & 127;
    if (i >= N_NODES) return;
    float di = dinv[i];
    float v = acc[(size_t)i * HIDDEN + f] + h[(size_t)i * HIDDEN + f] * di * di + bg[f];
    v = fmaxf(v, 0.0f);
    int g = batch[i];
    atomicAdd(&gsum[g * HIDDEN + f], v);
    if (f == 0) atomicAdd(&gcount[g], 1.0f);
}

// ---------------------------------------------------------------------------
// K7: MLP head + log_softmax. One block (128 thr) per graph; tiny GEMV in LDS.
// ---------------------------------------------------------------------------
__global__ void __launch_bounds__(128)
k_head(const float* __restrict__ gsum, const float* __restrict__ gcount,
       const float* __restrict__ W1, const float* __restrict__ b1,
       const float* __restrict__ W2, const float* __restrict__ b2,
       float* __restrict__ out) {
    __shared__ float sp[HIDDEN];
    __shared__ float sh1[HIDDEN];
    __shared__ float slog[OUT_DIM];
    int t = threadIdx.x, g = blockIdx.x;

    float cnt = fmaxf(gcount[g], 1.0f);
    sp[t] = gsum[g * HIDDEN + t] / cnt;
    __syncthreads();

    float a = b1[t];
#pragma unroll 8
    for (int k = 0; k < HIDDEN; ++k) a += sp[k] * W1[k * HIDDEN + t];
    sh1[t] = fmaxf(a, 0.0f);
    __syncthreads();

    if (t < OUT_DIM) {
        float l = b2[t];
#pragma unroll 8
        for (int j = 0; j < HIDDEN; ++j) l += sh1[j] * W2[j * OUT_DIM + t];
        slog[t] = l;
    }
    __syncthreads();
    if (t == 0) {
        float m   = fmaxf(slog[0], slog[1]);
        float lse = m + logf(expf(slog[0] - m) + expf(slog[1] - m));
        out[g * OUT_DIM + 0] = slog[0] - lse;
        out[g * OUT_DIM + 1] = slog[1] - lse;
    }
}

// ---------------------------------------------------------------------------
// host-side launcher (graph-capture safe: only kernel launches on `stream`)
// ---------------------------------------------------------------------------
static inline size_t align256(size_t x) { return (x + 255) & ~(size_t)255; }

extern "C" void kernel_launch(void* const* d_in, const int* in_sizes, int n_in,
                              void* d_out, int out_size, void* d_ws, size_t ws_size,
                              hipStream_t stream) {
    const float* x     = (const float*)d_in[0];
    const int*   ei    = (const int*)  d_in[1];   // [2, E] (src row, dst row)
    const int*   batch = (const int*)  d_in[2];   // [N]
    const float* Wg    = (const float*)d_in[3];   // [768,128]
    const float* bg    = (const float*)d_in[4];   // [128]
    const float* W1    = (const float*)d_in[5];   // [128,128]
    const float* b1    = (const float*)d_in[6];
    const float* W2    = (const float*)d_in[7];   // [128,2]
    const float* b2    = (const float*)d_in[8];
    float*       out   = (float*)d_out;           // [128,2]

    // ---- workspace layout (~50 MB) ----
    char*  ws  = (char*)d_ws;
    size_t off = 0;
    unsigned int* deg = (unsigned int*)(ws + off); off += align256(N_NODES * 4);
    float* dinv       = (float*)(ws + off);        off += align256(N_NODES * 4);
    unsigned short* whi = (unsigned short*)(ws + off); off += align256((size_t)IN_DIM * HIDDEN * 2);
    unsigned short* wlo = (unsigned short*)(ws + off); off += align256((size_t)IN_DIM * HIDDEN * 2);
    float* h          = (float*)(ws + off);        off += align256((size_t)N_NODES * HIDDEN * 4);
    float* acc        = (float*)(ws + off);        off += align256((size_t)N_NODES * HIDDEN * 4);
    float* gsum       = (float*)(ws + off);        off += align256((size_t)N_GRAPHS * HIDDEN * 4);
    float* gcount     = (float*)(ws + off);        off += align256(N_GRAPHS * 4);
    (void)ws_size; (void)in_sizes; (void)n_in; (void)out_size;

    // K0: init (covers acc = 6.4M floats)
    {
        int total = N_NODES * HIDDEN;
        k_init<<<(total + 255) / 256, 256, 0, stream>>>(deg, acc, gsum, gcount);
    }
    // K1: degrees
    k_degree<<<(N_EDGES + 255) / 256, 256, 0, stream>>>(ei, deg);
    // K2: deg^-1/2
    k_dinv<<<(N_NODES + 255) / 256, 256, 0, stream>>>(deg, dinv);
    // K3: pack W into WMMA-B bf16 hi/lo fragments
    k_packW<<<(IN_DIM * HIDDEN + 255) / 256, 256, 0, stream>>>(Wg, whi, wlo);
    // K4: h = x @ W  (WMMA bf16x3, LDS-staged A, double-buffered B)
    k_gemm_wmma<<<M_TILES, 256, 0, stream>>>(x, (const unsigned int*)whi,
                                             (const unsigned int*)wlo, h);
    // K5: edge message scatter
    k_scatter<<<(N_EDGES + 1) / 2, 256, 0, stream>>>(ei, h, dinv, acc);
    // K6: self loop + bias + relu + mean-pool accumulate
    k_relu_pool<<<(N_NODES + 1) / 2, 256, 0, stream>>>(acc, h, dinv, bg, batch,
                                                       gsum, gcount);
    // K7: MLP head + log_softmax
    k_head<<<N_GRAPHS, 128, 0, stream>>>(gsum, gcount, W1, b1, W2, b2, out);
}
